// HeteroGAT_23776938951053
// MI455X (gfx1250) — matrix operand order repaired and checked by reading, not compiled, not run on hardware
//
#include <hip/hip_runtime.h>
#include <hip/hip_bf16.h>

// ---------------------------------------------------------------------------
// HeteroGAT forward for MI455X (gfx1250), wave32, WMMA f32<-f16 16x16x32.
// GEMM stages the weight panel in LDS (pre-swizzled into WMMA fragment
// order) so the inner loop is ds_load_b128 + v_wmma, not strided vmem.
// ---------------------------------------------------------------------------

typedef __attribute__((ext_vector_type(16))) _Float16 v16h;
typedef __attribute__((ext_vector_type(8)))  float    v8f;

#define HID 64
#define HC  256   // H * C
#define NH  4

static inline int cdiv(long a, long b) { return (int)((a + b - 1) / b); }

// ---------------- WMMA GEMM: C[M,NCOL] = A[M,64] @ B[64,NCOL] (+bias) -------
// A,B,C are f32 in global; compute is f16 WMMA with f32 accum.
// Block = 256 threads (8 waves). Block owns 128 rows; wave w owns rows
// [blk*128 + 16w, +16). B (64 x NCOL) is staged once per block into LDS as
// f16, pre-swizzled so each lane's 16-element fragment is contiguous (32B).
// M must be a multiple of 16 (NC=50000, NP=20000 both are); the per-wave
// M-tile guard is wave-uniform so EXEC is all-ones around every WMMA.
__global__ void gat_gemm_k64_wmma(const float* __restrict__ A,
                                  const float* __restrict__ B,
                                  float* __restrict__ Cout,
                                  const float* __restrict__ bias,
                                  int NCOL, int M) {
  // Bs[(t*2 + kc)*32 + lane][16 halves]; max ntiles = 16 -> 32 KB
  __shared__ __align__(32) _Float16 Bs[16 * 2 * 32 * 16];

  const int lane = threadIdx.x & 31;
  const int wave = threadIdx.x >> 5;          // 0..7
  const int hi   = lane >> 4;                 // 0/1 half-wave
  const int ln   = lane & 15;
  const int ntiles = NCOL >> 4;               // 16 or 4

  // ---- stage B into LDS, converted to f16, in fragment order ----
  // B-fragment layout: lane n = 16*t + ln; j<8 -> k = kc*32 + 8*hi + j; j>=8 -> k = kc*32 + 16 + 8*hi + (j-8).
  for (int s = threadIdx.x; s < ntiles * 64; s += 256) {
    const int t    = s >> 6;
    const int kc   = (s >> 5) & 1;
    const int sl   = s & 31;                  // fragment lane
    const int shi  = sl >> 4, sln = sl & 15;
    const int n    = t * 16 + sln;
    const int kb   = kc * 32 + shi * 8;
    _Float16* dstp = &Bs[(size_t)s * 16];
#pragma unroll
    for (int j = 0; j < 16; ++j)
      dstp[j] = (_Float16)B[(long)(kb + j + (j >= 8 ? 8 : 0)) * NCOL + n];
  }
  __syncthreads();

  const int mtile = blockIdx.x * 8 + wave;    // wave-uniform
  if (mtile * 16 >= M) return;
  const int m0  = mtile * 16;
  const int row = m0 + ln;                    // A-matrix row for this lane

  // ---- A fragments for both K-chunks (K=64 -> 2 x 16x16x32) ----
  // ISA 16-bit A 16x32: lane m = ln; j<8 -> k = 8*hi + j; j>=8 -> k = 16+8*hi+(j-8).
  v16h afrag[2];
#pragma unroll
  for (int kc = 0; kc < 2; ++kc) {
    const float* ap = A + (long)row * HID + kc * 32 + hi * 8;
#pragma unroll
    for (int j = 0; j < 8; ++j) {
      afrag[kc][j]     = (_Float16)ap[j];
      afrag[kc][8 + j] = (_Float16)ap[16 + j];
    }
  }

  // ---- sweep column tiles: B from LDS, A in registers ----
  for (int t = 0; t < ntiles; ++t) {
    v8f acc = {};
#pragma unroll
    for (int kc = 0; kc < 2; ++kc) {
      const v16h bfrag = *(const v16h*)&Bs[(size_t)((t * 2 + kc) * 32 + lane) * 16];
      acc = __builtin_amdgcn_wmma_f32_16x16x32_f16(
          /*neg_a=*/false, afrag[kc], /*neg_b=*/false, bfrag,
          /*c_mod=*/(short)0, acc, /*reuse_a=*/false, /*reuse_b=*/false);
    }
    const int n0 = t << 4;
    const float badd = bias ? bias[n0 + ln] : 0.0f;
    // D layout: VGPR r -> M = r + 8*hi, N = ln.
#pragma unroll
    for (int r = 0; r < 8; ++r) {
      const int mm = m0 + r + 8 * hi;
      Cout[(long)mm * NCOL + n0 + ln] = acc[r] + badd;
    }
  }
}

// ---------------- encoder: h = relu(x[N,9] @ W[9,64] + b) -------------------
__global__ void gat_encoder(const float* __restrict__ x,
                            const float* __restrict__ W,
                            const float* __restrict__ b,
                            float* __restrict__ h, int N) {
  const long i = (long)blockIdx.x * blockDim.x + threadIdx.x;
  if (i >= (long)N * HID) return;
  const int n = (int)(i >> 6), c = (int)(i & 63);
  float acc = b[c];
#pragma unroll
  for (int k = 0; k < 9; ++k) acc += x[(long)n * 9 + k] * W[k * HID + c];
  h[i] = acc > 0.f ? acc : 0.f;
}

// ------- attention vector fold: u[k,h] = sum_c w[k,64h+c] * avec[h,c] ------
__global__ void gat_att_fold(const float* __restrict__ w,
                             const float* __restrict__ avec,
                             float* __restrict__ u) {
  const int i = threadIdx.x;           // 256 threads: k = i>>2, h = i&3
  const int k = i >> 2, hh = i & 3;
  float acc = 0.f;
#pragma unroll
  for (int c = 0; c < 64; ++c) acc += w[k * HC + hh * 64 + c] * avec[hh * 64 + c];
  u[i] = acc;
}

// ------------- node scores: a[n,h] = sum_k h[n,k] * u[k,h] ------------------
__global__ void gat_node_score(const float* __restrict__ hx,
                               const float* __restrict__ u,
                               float* __restrict__ out, int N) {
  const long i = (long)blockIdx.x * blockDim.x + threadIdx.x;
  if (i >= (long)N * NH) return;
  const int n = (int)(i >> 2), hh = (int)(i & 3);
  float acc = 0.f;
#pragma unroll
  for (int k = 0; k < 64; ++k) acc += hx[(long)n * HID + k] * u[k * NH + hh];
  out[i] = acc;
}

// order-preserving f32 <-> u32 for atomic max
__device__ __forceinline__ unsigned gat_enc(float f) {
  int b = __float_as_int(f);
  return (unsigned)(b ^ ((b >> 31) | 0x80000000));
}
__device__ __forceinline__ float gat_dec(unsigned u) {
  int b = (int)u;
  b = b ^ ((~b >> 31) | 0x80000000);
  return __int_as_float(b);
}

// ---- pass1: e = leaky(a_s[src]+a_d[dst]); segment max via atomicMax --------
__global__ void gat_edge_logit_max(const int* __restrict__ src,
                                   const int* __restrict__ dst,
                                   const float* __restrict__ a_s,
                                   const float* __restrict__ a_d,
                                   float* __restrict__ ebuf,
                                   unsigned* __restrict__ menc, int E) {
  const long i = (long)blockIdx.x * blockDim.x + threadIdx.x;
  if (i >= (long)E * NH) return;
  const int e = (int)(i >> 2), hh = (int)(i & 3);
  const int s = src[e], d = dst[e];
  float v = a_s[(long)s * NH + hh] + a_d[(long)d * NH + hh];
  v = v > 0.f ? v : 0.2f * v;           // leaky_relu, slope 0.2
  ebuf[i] = v;
  atomicMax(menc + (long)d * NH + hh, gat_enc(v));
}

// ---- pass2: ex = exp(e - m[dst]); denom[dst] += ex -------------------------
__global__ void gat_edge_exp_sum(const int* __restrict__ dst,
                                 const unsigned* __restrict__ menc,
                                 float* __restrict__ exbuf,   // in: e, out: ex
                                 float* __restrict__ denom, int E) {
  const long i = (long)blockIdx.x * blockDim.x + threadIdx.x;
  if (i >= (long)E * NH) return;
  const int e = (int)(i >> 2), hh = (int)(i & 3);
  const int d = dst[e];
  const float m  = gat_dec(menc[(long)d * NH + hh]);
  const float ex = __expf(exbuf[i] - m);
  exbuf[i] = ex;
  atomicAdd(denom + (long)d * NH + hh, ex);
}

// ---- pass3: agg[dst,c] += (1/H) * sum_h hs[src,h,c] * alpha[h] -------------
__global__ void gat_edge_scatter(const int* __restrict__ src,
                                 const int* __restrict__ dst,
                                 const float* __restrict__ hs,
                                 const float* __restrict__ exbuf,
                                 const float* __restrict__ denom,
                                 float* __restrict__ agg, int E) {
  const long i = (long)blockIdx.x * blockDim.x + threadIdx.x;
  if (i >= (long)E * 64) return;
  const int c = (int)(i & 63);
  const int e = (int)(i >> 6);
  const int s = src[e], d = dst[e];
  float acc = 0.f;
#pragma unroll
  for (int hh = 0; hh < NH; ++hh) {
    const float alpha = exbuf[(long)e * NH + hh] /
                        (denom[(long)d * NH + hh] + 1e-16f);
    acc += hs[(long)s * HC + hh * 64 + c] * alpha;
  }
  atomicAdd(agg + (long)d * HID + c, 0.25f * acc);
}

// ---- finalize: h = relu(agg + b1 (+ b2)) -----------------------------------
__global__ void gat_finalize(const float* __restrict__ agg,
                             const float* __restrict__ b1,
                             const float* __restrict__ b2,
                             float* __restrict__ hout, int N) {
  const long i = (long)blockIdx.x * blockDim.x + threadIdx.x;
  if (i >= (long)N * HID) return;
  const int c = (int)(i & 63);
  float v = agg[i] + b1[c] + (b2 ? b2[c] : 0.f);
  hout[i] = v > 0.f ? v : 0.f;
}

// ---------------------------------------------------------------------------
extern "C" void kernel_launch(void* const* d_in, const int* in_sizes, int n_in,
                              void* d_out, int out_size, void* d_ws, size_t ws_size,
                              hipStream_t stream) {
  (void)n_in; (void)out_size; (void)ws_size;

  // ---- inputs (setup_inputs() dict insertion order, nested in order) ----
  const float* x_car = (const float*)d_in[0];
  const float* x_ped = (const float*)d_in[1];
  const int*   e_cc  = (const int*)d_in[2];
  const int*   e_cp  = (const int*)d_in[3];
  const int*   e_pc  = (const int*)d_in[4];
  const int NC   = in_sizes[0] / 9;
  const int NP   = in_sizes[1] / 9;
  const int E_CC = in_sizes[2] / 2;
  const int E_CP = in_sizes[3] / 2;
  const int E_PC = in_sizes[4] / 2;

  const float* enc_car_W = (const float*)d_in[5];
  const float* enc_car_b = (const float*)d_in[6];
  const float* enc_ped_W = (const float*)d_in[7];
  const float* enc_ped_b = (const float*)d_in[8];
  const float* dec_car_W = (const float*)d_in[9];
  const float* dec_car_b = (const float*)d_in[10];
  const float* dec_ped_W = (const float*)d_in[11];
  const float* dec_ped_b = (const float*)d_in[12];
  // layer l, relation r (cc=0,cp=1,pc=2): base = 13 + 15*l + 5*r
  // fields: wsrc, wdst, asrc, adst, bias

  // ---- workspace carve-up ----
  char* wsp = (char*)d_ws;
  auto carve = [&](size_t bytes) -> void* {
    void* p = wsp; wsp += (bytes + 255) & ~(size_t)255; return p;
  };
  float* hc    = (float*)carve((size_t)NC * HID * 4);
  float* hp    = (float*)carve((size_t)NP * HID * 4);
  float* hc2   = (float*)carve((size_t)NC * HID * 4);
  float* hp2   = (float*)carve((size_t)NP * HID * 4);
  float* hsbuf = (float*)carve((size_t)NC * HC * 4);       // max src = NC
  float* a_s   = (float*)carve((size_t)NC * NH * 4);
  float* a_d   = (float*)carve((size_t)NC * NH * 4);
  unsigned* menc = (unsigned*)carve((size_t)NC * NH * 4);
  float* denom = (float*)carve((size_t)NC * NH * 4);
  int    E_MAX = E_CC > E_CP ? E_CC : E_CP; if (E_PC > E_MAX) E_MAX = E_PC;
  float* exbuf = (float*)carve((size_t)E_MAX * NH * 4);
  float* agg_c = (float*)carve((size_t)NC * HID * 4);
  float* agg_p = (float*)carve((size_t)NP * HID * 4);
  float* u_s   = (float*)carve(HID * NH * 4);
  float* u_d   = (float*)carve(HID * NH * 4);

  const int TB = 256;

  // ---- encoders ----
  gat_encoder<<<cdiv((long)NC * HID, TB), TB, 0, stream>>>(x_car, enc_car_W, enc_car_b, hc, NC);
  gat_encoder<<<cdiv((long)NP * HID, TB), TB, 0, stream>>>(x_ped, enc_ped_W, enc_ped_b, hp, NP);

  // ---- one GAT relation ----
  auto run_rel = [&](int pbase, const float* h_src, int Ns,
                     const float* h_dst, int Nd,
                     const int* edges, int E, float* agg) {
    const float* wsrc = (const float*)d_in[pbase + 0];
    const float* wdst = (const float*)d_in[pbase + 1];
    const float* asrc = (const float*)d_in[pbase + 2];
    const float* adst = (const float*)d_in[pbase + 3];
    const int* src = edges;
    const int* dst = edges + E;

    // hs = h_src @ wsrc   (WMMA, LDS-staged B)
    gat_gemm_k64_wmma<<<cdiv(Ns, 128), 256, 0, stream>>>(h_src, wsrc, hsbuf, nullptr, HC, Ns);
    // attention folds + node scores
    gat_att_fold<<<1, 256, 0, stream>>>(wsrc, asrc, u_s);
    gat_att_fold<<<1, 256, 0, stream>>>(wdst, adst, u_d);
    gat_node_score<<<cdiv((long)Ns * NH, TB), TB, 0, stream>>>(h_src, u_s, a_s, Ns);
    gat_node_score<<<cdiv((long)Nd * NH, TB), TB, 0, stream>>>(h_dst, u_d, a_d, Nd);
    // segment softmax + scatter
    hipMemsetAsync(menc,  0, (size_t)Nd * NH * 4, stream);
    hipMemsetAsync(denom, 0, (size_t)Nd * NH * 4, stream);
    gat_edge_logit_max<<<cdiv((long)E * NH, TB), TB, 0, stream>>>(src, dst, a_s, a_d, exbuf, menc, E);
    gat_edge_exp_sum<<<cdiv((long)E * NH, TB), TB, 0, stream>>>(dst, menc, exbuf, denom, E);
    gat_edge_scatter<<<cdiv((long)E * 64, TB), TB, 0, stream>>>(src, dst, hsbuf, exbuf, denom, agg, E);
  };

  float* cur_c = hc; float* cur_p = hp;
  float* nxt_c = hc2; float* nxt_p = hp2;

  for (int l = 0; l < 2; ++l) {
    const int b_cc = 13 + 15 * l + 0;
    const int b_cp = 13 + 15 * l + 5;
    const int b_pc = 13 + 15 * l + 10;
    hipMemsetAsync(agg_c, 0, (size_t)NC * HID * 4, stream);
    hipMemsetAsync(agg_p, 0, (size_t)NP * HID * 4, stream);
    run_rel(b_cc, cur_c, NC, cur_c, NC, e_cc, E_CC, agg_c);   // car -> car
    run_rel(b_pc, cur_p, NP, cur_c, NC, e_pc, E_PC, agg_c);   // ped -> car
    run_rel(b_cp, cur_c, NC, cur_p, NP, e_cp, E_CP, agg_p);   // car -> ped
    const float* bias_cc = (const float*)d_in[b_cc + 4];
    const float* bias_cp = (const float*)d_in[b_cp + 4];
    const float* bias_pc = (const float*)d_in[b_pc + 4];
    gat_finalize<<<cdiv((long)NC * HID, TB), TB, 0, stream>>>(agg_c, bias_cc, bias_pc, nxt_c, NC);
    gat_finalize<<<cdiv((long)NP * HID, TB), TB, 0, stream>>>(agg_p, bias_cp, nullptr,  nxt_p, NP);
    float* t;
    t = cur_c; cur_c = nxt_c; nxt_c = t;
    t = cur_p; cur_p = nxt_p; nxt_p = t;
  }

  // ---- decoders (WMMA, bias, no relu) straight into d_out ----
  float* out_c = (float*)d_out;
  float* out_p = out_c + (size_t)NC * HID;
  gat_gemm_k64_wmma<<<cdiv(NC, 128), 256, 0, stream>>>(cur_c, dec_car_W, out_c, dec_car_b, 64, NC);
  gat_gemm_k64_wmma<<<cdiv(NP, 128), 256, 0, stream>>>(cur_p, dec_ped_W, out_p, dec_ped_b, 64, NP);
}
